// RzLinear_17188459119213
// MI455X (gfx1250) — compile-verified, matrix operand order checked by reading, not run on hardware
//
#include <hip/hip_runtime.h>
#include <hip/hip_bf16.h>

// RzLinear (ROBE-Z compressed linear): out = X @ W, W gathered from a 4MB
// compressed buffer via gather_idx.
//
// MI455X strategy:
//  - Compute-bound (845 FLOP/byte) => bf16 K=32 WMMA path, bf16x3 split
//    (hi+lo) for ~fp32 accuracy.
//  - Stage 1 materializes W once into ws as transposed, k-tiled bf16 hi/lo
//    planes (64 MB ws) so every GEMM B-tile is one contiguous 16 KB block.
//  - Stage 2: 8-wave WG, 128x256 tile, 64x64 per wave. B tiles are staged by
//    the Tensor Data Mover (tensor_load_to_lds, TENSORcnt), double-buffered,
//    with TDM LDS padding (4 dwords per 16 dwords) recreating the 80B pitch
//    needed for conflict-free b128 fragment reads. A tiles are fp32->bf16
//    split in-register, loads issued before compute for latency hiding.

#define NDIM 8192
#define KDIM 4096
#define MDIM 4096

typedef __attribute__((ext_vector_type(16))) __bf16 v16bf;
typedef __attribute__((ext_vector_type(8)))  float  v8f;
typedef unsigned int u32x4 __attribute__((ext_vector_type(4)));
typedef int          i32x8 __attribute__((ext_vector_type(8)));
typedef int          i32x4 __attribute__((ext_vector_type(4)));

__device__ __forceinline__ unsigned short bfbits(__bf16 v) {
    union { __bf16 b; unsigned short u; } x; x.b = v; return x.u;
}

__device__ __forceinline__ void split_bf16(float x, unsigned short& hi, unsigned short& lo) {
    __bf16 h = (__bf16)x;          // RNE float->bf16
    float hf = (float)h;
    __bf16 l = (__bf16)(x - hf);   // residual
    hi = bfbits(h);
    lo = bfbits(l);
}

// Generic pointers to LDS carry the LDS byte offset in their low 32 bits
// (flat aperture).
__device__ __forceinline__ unsigned lds_off(const void* p) {
    return (unsigned)(uintptr_t)p;
}

// ---------------------------------------------------------------------------
// TDM: 1-D load of `ndw` dwords from gsrc into LDS at lds_addr, inserting a
// 4-dword pad after every 16 dwords (64B data + 16B pad = 80B pitch).
// D# layout per ISA 8.3/8.4: group0 = {flags, lds_addr, gaddr_lo, gaddr_hi|type},
// group1 = {dsize/pad, dim0_lo<<16, dim0_hi, tile0<<16, 0, stride0, 0, 0}.
// This toolchain exposes the 6-arg builtin: (g0, g1, g2, g3, g4, cpol).
// ---------------------------------------------------------------------------
__device__ __forceinline__ void tdm_load_1d_padded(const void* gsrc,
                                                   unsigned lds_addr,
                                                   int ndw) {
    unsigned long long ga = (unsigned long long)(uintptr_t)gsrc;
    u32x4 g0;
    g0[0] = 1u;                                   // count=1 (valid), user mode
    g0[1] = lds_addr;                             // lds_addr [63:32]
    g0[2] = (unsigned)(ga & 0xffffffffu);         // global_addr [95:64]
    g0[3] = (unsigned)((ga >> 32) & 0x01ffffffu)  // global_addr [120:96]
          | (2u << 30);                           // type=2 ("image")
    i32x8 g1;
    g1[0] = (int)((2u << 16)      // data_size = 4B
                | (1u << 20)      // pad_enable
                | (3u << 22)      // pad_interval: 16 dwords
                | (3u << 25));    // pad_amount: 4 dwords
    g1[1] = (int)(((unsigned)ndw & 0xffffu) << 16);   // tensor_dim0[15:0]
    g1[2] = (int)((unsigned)ndw >> 16);               // tensor_dim0[31:16]
    g1[3] = (int)(((unsigned)ndw & 0xffffu) << 16);   // tile_dim0
    g1[4] = 0;                                        // tile_dim1/2 = 0 (1-D)
    g1[5] = ndw;                                      // tensor_dim0_stride
    g1[6] = 0;
    g1[7] = 0;
    i32x4 z4 = {0, 0, 0, 0};                          // groups 2/3 unused (1-D)
    i32x8 z8 = {0, 0, 0, 0, 0, 0, 0, 0};
    __builtin_amdgcn_tensor_load_to_lds(g0, g1, z4, z4, z8, 0);
}

// ---------------------------------------------------------------------------
// Stage 1: gather + split + transpose into k-tiled layout.
// Output element (m, k) of W^T lives at:  (k/32)*(MDIM*32) + m*32 + (k%32)
// so each GEMM B-tile (256 m x 32 k) is one contiguous 16 KB block per plane.
// ---------------------------------------------------------------------------
__global__ __launch_bounds__(256) void gather_split_kernel(
    const float* __restrict__ hw,          // [1<<20] compressed weights
    const int*   __restrict__ gidx,        // [KDIM*MDIM]
    unsigned short* __restrict__ whi,      // bf16 hi plane (tiled)
    unsigned short* __restrict__ wlo)      // bf16 lo plane (tiled)
{
    __shared__ unsigned short shi[64 * 65];   // 64k x 64m tile, pitch 65 (u16)
    __shared__ unsigned short slo[64 * 65];

    const int t  = threadIdx.x;
    const int k0 = blockIdx.x * 64;
    const int m0 = blockIdx.y * 64;

    #pragma unroll
    for (int i = 0; i < 16; ++i) {
        int e = t + i * 256;
        int k = e >> 6;
        int m = e & 63;
        int gi = gidx[(k0 + k) * MDIM + (m0 + m)];
        float w = hw[gi];
        unsigned short h, l;
        split_bf16(w, h, l);
        shi[k * 65 + m] = h;
        slo[k * 65 + m] = l;
    }
    __syncthreads();

    #pragma unroll
    for (int r = 0; r < 2; ++r) {
        int m     = (t >> 3) + r * 32;
        int kbase = (t & 7) * 8;
        union { unsigned short s[8]; uint4 q; } ph, pl;
        #pragma unroll
        for (int j = 0; j < 8; ++j) {
            ph.s[j] = shi[(kbase + j) * 65 + m];
            pl.s[j] = slo[(kbase + j) * 65 + m];
        }
        int kk   = k0 + kbase;
        int addr = (kk >> 5) * (MDIM * 32) + (m0 + m) * 32 + (kk & 31);
        *(uint4*)(whi + addr) = ph.q;
        *(uint4*)(wlo + addr) = pl.q;
    }
}

// ---------------------------------------------------------------------------
// Stage 2: bf16x3 WMMA GEMM with TDM-staged, double-buffered B tiles.
// ---------------------------------------------------------------------------
union FragBF { v16bf v; uint4 q[2]; };

__global__ __launch_bounds__(256) void rz_gemm_kernel(
    const float* __restrict__ A,             // [NDIM][KDIM] fp32
    const unsigned short* __restrict__ bhiG, // tiled W^T hi plane
    const unsigned short* __restrict__ bloG, // tiled W^T lo plane
    float* __restrict__ out)                 // [NDIM][MDIM] fp32
{
    // pitch = 5 uint4 = 80 bytes per 32-k row (TDM pad recreates this for B)
    __shared__ uint4 aHi[128 * 5];
    __shared__ uint4 aLo[128 * 5];
    __shared__ uint4 bHi[2][256 * 5];
    __shared__ uint4 bLo[2][256 * 5];

    const int t    = threadIdx.x;
    const int lane = t & 31;
    const int w    = t >> 5;
    const int wn   = (w >> 2) * 64;
    const int wm   = (w & 3) * 64;

    const int n0 = blockIdx.y * 128;
    const int m0 = blockIdx.x * 256;

    v8f c[4][4];
    const v8f vzero = {0.f, 0.f, 0.f, 0.f, 0.f, 0.f, 0.f, 0.f};
    #pragma unroll
    for (int i = 0; i < 4; ++i)
        #pragma unroll
        for (int j = 0; j < 4; ++j)
            c[i][j] = vzero;

    const int ar = t >> 1;            // A staging: 2 threads/row, 16 floats
    const int ak = (t & 1) * 16;
    const int abase = ar * 5 + (ak >> 3);
    const float* aRow = A + (n0 + ar) * KDIM + ak;

    const int lrow = lane & 15;
    const int loff = lane >> 4;

    const unsigned ldsBHi[2] = { lds_off(&bHi[0][0]), lds_off(&bHi[1][0]) };
    const unsigned ldsBLo[2] = { lds_off(&bLo[0][0]), lds_off(&bLo[1][0]) };

    // ---- prologue: TDM B(0) -> buf0, stage A(0) ----
    if (w == 0) {
        const int src0 = (m0) * 32;   // kt = 0
        tdm_load_1d_padded(bhiG + src0, ldsBHi[0], 256 * 32 / 2);
        tdm_load_1d_padded(bloG + src0, ldsBLo[0], 256 * 32 / 2);
    }
    {
        union { unsigned short s[16]; uint4 q[2]; } ph, pl;
        #pragma unroll
        for (int q4 = 0; q4 < 4; ++q4) {
            float4 f = ((const float4*)aRow)[q4];
            float xs[4] = {f.x, f.y, f.z, f.w};
            #pragma unroll
            for (int e = 0; e < 4; ++e)
                split_bf16(xs[e], ph.s[q4 * 4 + e], pl.s[q4 * 4 + e]);
        }
        aHi[abase] = ph.q[0]; aHi[abase + 1] = ph.q[1];
        aLo[abase] = pl.q[0]; aLo[abase + 1] = pl.q[1];
    }
    if (w == 0) __builtin_amdgcn_s_wait_tensorcnt(0);
    __syncthreads();

    for (int kt = 0; kt < KDIM / 32; ++kt) {
        const int cur = kt & 1;
        const int nxt = cur ^ 1;
        const bool more = (kt + 1) < (KDIM / 32);

        // issue next B tile via TDM + next A global loads before compute
        float areg[16];
        if (more) {
            if (w == 0) {
                const int src = (kt + 1) * (MDIM * 32) + m0 * 32;
                tdm_load_1d_padded(bhiG + src, ldsBHi[nxt], 256 * 32 / 2);
                tdm_load_1d_padded(bloG + src, ldsBLo[nxt], 256 * 32 / 2);
            }
            const float* ap = aRow + (kt + 1) * 32;
            #pragma unroll
            for (int q4 = 0; q4 < 4; ++q4) {
                float4 f = ((const float4*)ap)[q4];
                areg[q4 * 4 + 0] = f.x; areg[q4 * 4 + 1] = f.y;
                areg[q4 * 4 + 2] = f.z; areg[q4 * 4 + 3] = f.w;
            }
        }

        // ---- compute current K-step ----
        FragBF ah[4], al[4];
        #pragma unroll
        for (int i = 0; i < 4; ++i) {
            int base = (wn + i * 16 + lrow) * 5 + loff;
            ah[i].q[0] = aHi[base];     ah[i].q[1] = aHi[base + 2];
            al[i].q[0] = aLo[base];     al[i].q[1] = aLo[base + 2];
        }
        #pragma unroll
        for (int j = 0; j < 4; ++j) {
            int bb = (wm + j * 16 + lrow) * 5 + loff;
            FragBF bh, bl;
            bh.q[0] = bHi[cur][bb];  bh.q[1] = bHi[cur][bb + 2];
            bl.q[0] = bLo[cur][bb];  bl.q[1] = bLo[cur][bb + 2];
            #pragma unroll
            for (int i = 0; i < 4; ++i) {
                c[i][j] = __builtin_amdgcn_wmma_f32_16x16x32_bf16(
                    false, ah[i].v, false, bh.v, (short)0, c[i][j], false, false);
                c[i][j] = __builtin_amdgcn_wmma_f32_16x16x32_bf16(
                    false, ah[i].v, false, bl.v, (short)0, c[i][j], false, false);
                c[i][j] = __builtin_amdgcn_wmma_f32_16x16x32_bf16(
                    false, al[i].v, false, bh.v, (short)0, c[i][j], false, false);
            }
        }
        __syncthreads();   // A/B reads of this step complete

        // ---- publish next A tile (single-buffered, safe after barrier) ----
        if (more) {
            union { unsigned short s[16]; uint4 q[2]; } ph, pl;
            #pragma unroll
            for (int e = 0; e < 16; ++e)
                split_bf16(areg[e], ph.s[e], pl.s[e]);
            aHi[abase] = ph.q[0]; aHi[abase + 1] = ph.q[1];
            aLo[abase] = pl.q[0]; aLo[abase + 1] = pl.q[1];
            if (w == 0) __builtin_amdgcn_s_wait_tensorcnt(0);
        }
        __syncthreads();   // next A + TDM B visible to all waves
    }

    // ---- writeback ----
    #pragma unroll
    for (int i = 0; i < 4; ++i) {
        int nb = n0 + wn + i * 16 + loff * 8;
        #pragma unroll
        for (int j = 0; j < 4; ++j) {
            int mb = m0 + wm + j * 16 + lrow;
            #pragma unroll
            for (int r = 0; r < 8; ++r)
                out[(nb + r) * MDIM + mb] = c[i][j][r];
        }
    }
}

extern "C" void kernel_launch(void* const* d_in, const int* in_sizes, int n_in,
                              void* d_out, int out_size, void* d_ws, size_t ws_size,
                              hipStream_t stream) {
    const float* hw  = (const float*)d_in[0];   // hashed_weights [1<<20]
    const float* x   = (const float*)d_in[1];   // input_v [8192][4096]
    const int*   gix = (const int*)d_in[2];     // gather_idx [4096*4096]
    float* out = (float*)d_out;

    unsigned short* whi = (unsigned short*)d_ws;            // 32 MB
    unsigned short* wlo = whi + (size_t)KDIM * MDIM;        // 32 MB

    gather_split_kernel<<<dim3(KDIM / 64, MDIM / 64), 256, 0, stream>>>(hw, gix, whi, wlo);
    rz_gemm_kernel<<<dim3(MDIM / 256, NDIM / 128), 256, 0, stream>>>(x, whi, wlo, out);
}
